// Model_83116207112285
// MI455X (gfx1250) — matrix operand order repaired and checked by reading, not compiled, not run on hardware
//
#include <hip/hip_runtime.h>
#include <hip/hip_bf16.h>
#include <math.h>

namespace {

constexpr int Bn = 8, Ln = 512, Cn = 321;
constexpr int Mrows = Bn * Cn;   // 2568
constexpr int N1 = 512, N2 = 128, PRED = 96;

typedef _Float16 v16h __attribute__((ext_vector_type(16)));
typedef _Float16 v8h  __attribute__((ext_vector_type(8)));
typedef float    v8f  __attribute__((ext_vector_type(8)));

__device__ __forceinline__ float siluf(float x) { return x / (1.f + expf(-x)); }
__device__ __forceinline__ float softplusf(float x) {
  if (x > 20.f) return x;
  if (x < -20.f) return expf(x);
  return log1pf(expf(x));
}

// ------------------------------------------------------------------ utility
__global__ __launch_bounds__(256) void f32_to_f16_kernel(const float* __restrict__ in,
                                                         _Float16* __restrict__ out, int n) {
  int i = blockIdx.x * 256 + threadIdx.x;
  if (i < n) out[i] = (_Float16)in[i];
}

__global__ __launch_bounds__(256) void add3_kernel(const float* __restrict__ a,
    const float* __restrict__ b, const float* __restrict__ c,
    float* __restrict__ o, int n) {
  int i = blockIdx.x * 256 + threadIdx.x;
  if (i < n) o[i] = a[i] + b[i] + c[i];
}

// ------------------------------------------------- RevIN stats + transpose
// xnT[m][l] with m = b*C + c ; also records mean/std per (b,c)
__global__ __launch_bounds__(256) void revin_kernel(const float* __restrict__ x,
    const float* __restrict__ rw, const float* __restrict__ rb,
    float* __restrict__ xnT, _Float16* __restrict__ xnTh,
    float* __restrict__ meanv, float* __restrict__ stdv) {
  int m = blockIdx.x;
  int b = m / Cn, c = m % Cn;
  int tid = threadIdx.x;
  __shared__ float red[256];
  float s = 0.f, sq = 0.f;
  for (int l = tid; l < Ln; l += 256) {
    float v = x[((size_t)b * Ln + l) * Cn + c];
    s += v; sq += v * v;
  }
  red[tid] = s; __syncthreads();
  for (int o = 128; o > 0; o >>= 1) { if (tid < o) red[tid] += red[tid + o]; __syncthreads(); }
  float mean = red[0] / Ln;
  __syncthreads();
  red[tid] = sq; __syncthreads();
  for (int o = 128; o > 0; o >>= 1) { if (tid < o) red[tid] += red[tid + o]; __syncthreads(); }
  float var = red[0] / Ln - mean * mean;
  float stdd = sqrtf(var + 1e-5f);
  if (tid == 0) { meanv[m] = mean; stdv[m] = stdd; }
  float g = rw[c], be = rb[c];
  float inv = 1.f / stdd;
  for (int l = tid; l < Ln; l += 256) {
    float v = x[((size_t)b * Ln + l) * Cn + c];
    float xn = (v - mean) * inv * g + be;
    xnT[(size_t)m * Ln + l] = xn;
    xnTh[(size_t)m * Ln + l] = (_Float16)xn;
  }
}

// --------------------------------------------------------------- WMMA GEMM
// C[M,N] = A[M,K] * W[N,K]^T (+bias) ; one wave per 16x16 C-tile, 8 waves/block.
// A fragment (16-bit A 16x32):            lane<16 -> K 0..15, lane>=16 -> K 16..31
// B fragment (16-bit B 32x16):            lane<16 -> K 0..15, lane>=16 -> K 16..31
__global__ __launch_bounds__(256) void wmma_gemm_f16(
    const _Float16* __restrict__ A,   // M x K row-major
    const _Float16* __restrict__ W,   // N x K row-major ("oi" weight)
    const float* __restrict__ bias,   // N or null
    float* __restrict__ outF, int ldF,
    _Float16* __restrict__ outH, int ldH,
    int M, int N, int K, int act) {
  const int lane = threadIdx.x & 31;
  const int wave = threadIdx.x >> 5;
  const int nTile = blockIdx.x;
  const int mTile = blockIdx.y * 8 + wave;
  if (mTile * 16 >= M) return;                    // wave-uniform exit (EXEC all-1 kept)
  int rowA = mTile * 16 + (lane & 15); if (rowA > M - 1) rowA = M - 1;
  int rowW = nTile * 16 + (lane & 15); if (rowW > N - 1) rowW = N - 1;
  const int kbA = (lane < 16) ? 0 : 16;
  const int kbB = (lane < 16) ? 0 : 16;
  const _Float16* Ap = A + (size_t)rowA * K;
  const _Float16* Wp = W + (size_t)rowW * K;
  v8f acc = {};
  for (int k0 = 0; k0 < K; k0 += 32) {
    union { v16h v; v8h p[2]; } af, bf;
    af.p[0] = *(const v8h*)(Ap + k0 + kbA);
    af.p[1] = *(const v8h*)(Ap + k0 + kbA + 8);
    bf.v    = *(const v16h*)(Wp + k0 + kbB);
    acc = __builtin_amdgcn_wmma_f32_16x16x32_f16(false, af.v, false, bf.v,
                                                 (short)0, acc, false, false);
  }
  const int col = nTile * 16 + (lane & 15);
  const float bv = (bias != nullptr && col < N) ? bias[col] : 0.f;
  const int rbase = mTile * 16 + ((lane < 16) ? 0 : 8);
#pragma unroll
  for (int r = 0; r < 8; ++r) {
    int row = rbase + r;
    if (row < M && col < N) {
      float v = acc[r] + bv;
      if (act) v = fmaxf(v, 0.f);
      if (outF) outF[(size_t)row * ldF + col] = v;
      if (outH) outH[(size_t)row * ldH + col] = (_Float16)v;
    }
  }
}

// -------------------------------------- series_decomp (t-dim==1) + circular corr
// trend = moving-mean of 25 identical samples = (25h)/25 ; res = h - trend (~0)
// st[t] = sum_tau res[(tau+t) mod n] * trend[tau]   (= irfft(S * conj(T)))
__global__ __launch_bounds__(256) void decomp_corr_kernel(const float* __restrict__ h,
    _Float16* __restrict__ sth, int n) {
  __shared__ float sres[512];
  __shared__ float strd[512];
  int m = blockIdx.x;
  int tid = threadIdx.x;
  for (int i = tid; i < n; i += 256) {
    float v = h[(size_t)m * n + i];
    float mm = (v * 25.f) / 25.f;
    sres[i] = v - mm;
    strd[i] = mm;
  }
  __syncthreads();
  for (int t = tid; t < n; t += 256) {
    float acc = 0.f;
    for (int tau = 0; tau < n; ++tau) {
      int idx = tau + t; if (idx >= n) idx -= n;
      acc += sres[idx] * strd[tau];
    }
    sth[(size_t)m * n + t] = (_Float16)acc;
  }
}

// ------------------------------------------------------ mamba2 (seq len == 1)
__global__ __launch_bounds__(256) void m2_convsilu_kernel(const float* __restrict__ xz,
    const float* __restrict__ convw, const float* __restrict__ convb,
    float* __restrict__ xs, _Float16* __restrict__ xsh, int di) {
  int i = blockIdx.x * 256 + threadIdx.x;
  if (i >= Mrows * di) return;
  int m = i / di, d = i - m * di;
  float xv = xz[(size_t)m * (2 * di) + d];
  float v = siluf(xv * convw[d * 2 + 1] + convb[d]);   // causal conv k=2, left pad 0
  xs[i] = v;
  xsh[i] = (_Float16)v;
}

__global__ __launch_bounds__(256) void m2_dt_kernel(const float* __restrict__ xdbl,
    const float* __restrict__ dtw, const float* __restrict__ dtb,
    float* __restrict__ dt, int di, int r, int xd) {
  int i = blockIdx.x * 256 + threadIdx.x;
  if (i >= Mrows * di) return;
  int m = i / di, d = i - m * di;
  const float* xrow = xdbl + (size_t)m * xd;
  float acc = dtb[d];
  for (int j = 0; j < r; ++j) acc += xrow[j] * dtw[d * r + j];
  dt[i] = softplusf(acc);
}

// h0=0, L=1  =>  y = dt*xs*sum_n(B_n*C_n) + xs*D ; then gated by silu(z)
__global__ __launch_bounds__(256) void m2_y_kernel(const float* __restrict__ xz,
    const float* __restrict__ xs, const float* __restrict__ dt,
    const float* __restrict__ xdbl, const float* __restrict__ Dv,
    _Float16* __restrict__ yh, int di, int r, int xd) {
  int i = blockIdx.x * 256 + threadIdx.x;
  if (i >= Mrows * di) return;
  int m = i / di, d = i - m * di;
  const float* xrow = xdbl + (size_t)m * xd;
  float S = 0.f;
#pragma unroll
  for (int nn = 0; nn < 16; ++nn) S += xrow[r + nn] * xrow[r + 16 + nn];
  float xsv = xs[i];
  float z = xz[(size_t)m * (2 * di) + di + d];
  float y = (dt[i] * xsv * S + xsv * Dv[d]) * siluf(z);
  yh[i] = (_Float16)y;
}

// ----------------------------- mamba1: d_model=1, d_inner=1, n=16, scan over S
// one row per 16-lane half-wave; lane = state index; shfl_xor tree reduction
__global__ __launch_bounds__(256) void mamba1_scan_kernel(const float* __restrict__ hseq,
    float* __restrict__ out, int S,
    const float* __restrict__ inproj, const float* __restrict__ convw,
    const float* __restrict__ convb, const float* __restrict__ xproj,
    const float* __restrict__ dtw, const float* __restrict__ dtb,
    const float* __restrict__ Alog, const float* __restrict__ Dv,
    const float* __restrict__ outproj) {
  int half = threadIdx.x >> 4;     // 16 rows per block
  int n = threadIdx.x & 15;        // state index
  int m = blockIdx.x * 16 + half;
  int mr = m < Mrows ? m : Mrows - 1;
  const float inp0 = inproj[0], inp1 = inproj[1];
  const float w0 = convw[0], w1 = convw[1], cb = convb[0];
  const float xp0 = xproj[0], Bc = xproj[1 + n], Cc = xproj[17 + n];
  const float dtwv = dtw[0], dtbv = dtb[0], D0 = Dv[0], ow = outproj[0];
  const float A = -expf(Alog[n]);
  float hstate = 0.f, prev = 0.f;
  const float* hp = hseq + (size_t)mr * S;
  float* op = out + (size_t)mr * S;
  for (int s = 0; s < S; ++s) {
    float u = hp[s];
    float xr = inp0 * u;
    float z = inp1 * u;
    float xc = prev * w0 + xr * w1 + cb;
    prev = xr;
    float xsv = siluf(xc);
    float dtv = softplusf(dtwv * (xp0 * xsv) + dtbv);
    float dA = expf(dtv * A);
    hstate = dA * hstate + dtv * (Bc * xsv) * xsv;
    float yn = hstate * (Cc * xsv);
    yn += __shfl_xor(yn, 1, 32);
    yn += __shfl_xor(yn, 2, 32);
    yn += __shfl_xor(yn, 4, 32);
    yn += __shfl_xor(yn, 8, 32);
    float y = (yn + xsv * D0) * siluf(z);
    if (m < Mrows && n == 0) op[s] = y * ow;
  }
}

// ------------------------------------------------------------- layernorm
// t = A (+Badd);  out = (t-mean)*rstd*g + be (+ t if addback)
__global__ __launch_bounds__(256) void layernorm_kernel(const float* __restrict__ A,
    const float* __restrict__ Badd, const float* __restrict__ g,
    const float* __restrict__ be, float* __restrict__ outF,
    _Float16* __restrict__ outH, int D, int ldH, int addback) {
  int m = blockIdx.x;
  int tid = threadIdx.x;
  __shared__ float red[256];
  float s = 0.f, sq = 0.f;
  for (int d = tid; d < D; d += 256) {
    float t = A[(size_t)m * D + d];
    if (Badd) t += Badd[(size_t)m * D + d];
    s += t; sq += t * t;
  }
  red[tid] = s; __syncthreads();
  for (int o = 128; o > 0; o >>= 1) { if (tid < o) red[tid] += red[tid + o]; __syncthreads(); }
  float mean = red[0] / D;
  __syncthreads();
  red[tid] = sq; __syncthreads();
  for (int o = 128; o > 0; o >>= 1) { if (tid < o) red[tid] += red[tid + o]; __syncthreads(); }
  float rstd = rsqrtf(red[0] / D - mean * mean + 1e-5f);
  for (int d = tid; d < D; d += 256) {
    float t = A[(size_t)m * D + d];
    if (Badd) t += Badd[(size_t)m * D + d];
    float v = (t - mean) * rstd * g[d] + be[d];
    if (addback) v += t;
    if (outF) outF[(size_t)m * D + d] = v;
    if (outH) outH[(size_t)m * ldH + d] = (_Float16)v;
  }
}

// ------------------------------------------------------------ de-normalize
__global__ __launch_bounds__(256) void final_kernel(const float* __restrict__ outp,
    const float* __restrict__ rw, const float* __restrict__ rb,
    const float* __restrict__ meanv, const float* __restrict__ stdv,
    float* __restrict__ out) {
  int idx = blockIdx.x * 256 + threadIdx.x;
  if (idx >= Bn * PRED * Cn) return;
  int c = idx % Cn;
  int p = (idx / Cn) % PRED;
  int b = idx / (Cn * PRED);
  int m = b * Cn + c;
  float v = outp[(size_t)m * PRED + p];
  out[idx] = (v - rb[c]) / (rw[c] + 1e-10f) * stdv[m] + meanv[m];
}

} // anonymous namespace

// input flattening: setup_inputs() dict insertion order, depth-first
enum {
  IN_X = 0, IN_REVW, IN_REVB,
  B1_LIN1W, B1_LIN1B,
  B1M1_INPROJ, B1M1_CONVW, B1M1_CONVB, B1M1_XPROJ, B1M1_DTW, B1M1_DTB, B1M1_ALOG, B1M1_D, B1M1_OUTPROJ,
  B1M2_INPROJ, B1M2_CONVW, B1M2_CONVB, B1M2_XPROJ, B1M2_DTW, B1M2_DTB, B1M2_ALOG, B1M2_D, B1M2_OUTPROJ,
  B2_LIN1W, B2_LIN1B,
  B2M1_INPROJ, B2M1_CONVW, B2M1_CONVB, B2M1_XPROJ, B2M1_DTW, B2M1_DTB, B2M1_ALOG, B2M1_D, B2M1_OUTPROJ,
  B2M2_INPROJ, B2M2_CONVW, B2M2_CONVB, B2M2_XPROJ, B2M2_DTW, B2M2_DTB, B2M2_ALOG, B2M2_D, B2M2_OUTPROJ,
  B2_LIN2W, B2_LIN2B,
  LN_G, LN_B, LN2_G, LN2_B,
  FD1_W1, FD1_B1, FD1_W2, FD1_B2,
  FD2_W1, FD2_B1, FD2_W2, FD2_B2,
  LIN3_W, LIN3_B, LIN4_W, LIN4_B,
  N_INPUTS
};

extern "C" void kernel_launch(void* const* d_in, const int* in_sizes, int n_in,
                              void* d_out, int out_size, void* d_ws, size_t ws_size,
                              hipStream_t stream) {
  (void)in_sizes; (void)out_size; (void)ws_size;
  const float* F[N_INPUTS] = {};
  for (int i = 0; i < n_in && i < N_INPUTS; ++i) F[i] = (const float*)d_in[i];

  char* base = (char*)d_ws;
  size_t off = 0;
  auto alloc = [&](size_t bytes) -> void* {
    void* p = base + off;
    off = (off + bytes + 255) & ~(size_t)255;
    return p;
  };
  auto af = [&](size_t n) { return (float*)alloc(n * 4); };
  auto ah = [&](size_t n) { return (_Float16*)alloc(n * 2); };

  const int M = Mrows;
  // f32 activations
  float* meanv = af(M);            float* stdv  = af(M);
  float* xnT   = af((size_t)M*N1); float* h1    = af((size_t)M*N1);
  float* g2    = af((size_t)M*N1); float* h2    = af((size_t)M*N2);
  float* xz1   = af((size_t)M*1024); float* xz2 = af((size_t)M*256);
  float* xs1   = af((size_t)M*512);  float* xs2 = af((size_t)M*128);
  float* xdbl1 = af((size_t)M*65);   float* xdbl2 = af((size_t)M*40);
  float* dt1   = af((size_t)M*512);  float* dt2 = af((size_t)M*128);
  float* m2o1  = af((size_t)M*512);  float* m2o2 = af((size_t)M*128);
  float* m1o1  = af((size_t)M*512);  float* m1o2 = af((size_t)M*128);
  float* xo1   = af((size_t)M*512);  float* xo2 = af((size_t)M*128);
  float* x1a   = af((size_t)M*512);  float* x2a = af((size_t)M*128);
  float* f1out = af((size_t)M*512);  float* f2out = af((size_t)M*128);
  float* outp  = af((size_t)M*PRED);
  // f16 activations
  _Float16* xnTh = ah((size_t)M*N1); _Float16* g2h  = ah((size_t)M*N1);
  _Float16* st1h = ah((size_t)M*512); _Float16* st2h = ah((size_t)M*128);
  _Float16* xs1h = ah((size_t)M*512); _Float16* xs2h = ah((size_t)M*128);
  _Float16* y1h  = ah((size_t)M*512); _Float16* y2h  = ah((size_t)M*128);
  _Float16* x1ah = ah((size_t)M*512); _Float16* x2ah = ah((size_t)M*128);
  _Float16* ffh  = ah((size_t)M*2048);          // reused by both FFN blocks
  _Float16* x2bh = ah((size_t)M*128);
  _Float16* cat16 = ah((size_t)M*1024);         // [lin3(x2) | x1]
  // f16 weights
  _Float16* lin1w1h = ah(512*512); _Float16* lin1w2h = ah(512*512);
  _Float16* lin2wh  = ah(128*512);
  _Float16* inproj1h = ah(1024*512); _Float16* xproj1h = ah(65*512);
  _Float16* outproj1h = ah(512*512);
  _Float16* inproj2h = ah(256*128);  _Float16* xproj2h = ah(40*128);
  _Float16* outproj2h = ah(128*128);
  _Float16* fd1w1h = ah(2048*512); _Float16* fd1w2h = ah(512*2048);
  _Float16* fd2w1h = ah(2048*128); _Float16* fd2w2h = ah(128*2048);
  _Float16* lin3h = ah(512*128);   _Float16* lin4h = ah(96*1024);

  auto toh = [&](const float* in, _Float16* out, int n) {
    f32_to_f16_kernel<<<(n + 255) / 256, 256, 0, stream>>>(in, out, n);
  };
  auto gemm = [&](const _Float16* A, const _Float16* W, const float* bias,
                  float* outF, int ldF, _Float16* outH, int ldH,
                  int Mm, int Nn, int Kk, int act) {
    dim3 grid((Nn + 15) / 16, (((Mm + 15) / 16) + 7) / 8);
    wmma_gemm_f16<<<grid, dim3(256), 0, stream>>>(A, W, bias, outF, ldF, outH, ldH,
                                                  Mm, Nn, Kk, act);
  };
  auto ewg = [](int n) { return dim3((n + 255) / 256); };
  auto ln = [&](const float* A, const float* Badd, const float* g, const float* be,
                float* outF, _Float16* outH, int D, int ldH, int addback) {
    layernorm_kernel<<<M, 256, 0, stream>>>(A, Badd, g, be, outF, outH, D, ldH, addback);
  };
  auto run_mamba2 = [&](const _Float16* sth, int dm, int di, int r, int xd,
                        const _Float16* inprojh, const float* convw, const float* convb,
                        const _Float16* xprojh, const float* dtw, const float* dtb,
                        const float* Dv, const _Float16* outprojh,
                        float* xz, float* xs, _Float16* xsh, float* xdbl,
                        float* dt, _Float16* yh, float* m2o) {
    gemm(sth, inprojh, nullptr, xz, 2 * di, nullptr, 0, M, 2 * di, dm, 0);
    m2_convsilu_kernel<<<ewg(M * di), 256, 0, stream>>>(xz, convw, convb, xs, xsh, di);
    gemm(xsh, xprojh, nullptr, xdbl, xd, nullptr, 0, M, xd, di, 0);
    m2_dt_kernel<<<ewg(M * di), 256, 0, stream>>>(xdbl, dtw, dtb, dt, di, r, xd);
    m2_y_kernel<<<ewg(M * di), 256, 0, stream>>>(xz, xs, dt, xdbl, Dv, yh, di, r, xd);
    gemm(yh, outprojh, nullptr, m2o, dm, nullptr, 0, M, dm, di, 0);
  };
  auto run_mamba1 = [&](const float* hseq, float* out, int S, int ib) {
    mamba1_scan_kernel<<<(M + 15) / 16, 256, 0, stream>>>(hseq, out, S,
        F[ib + 0], F[ib + 1], F[ib + 2], F[ib + 3], F[ib + 4], F[ib + 5],
        F[ib + 6], F[ib + 7], F[ib + 8]);
  };

  // ---- weight conversions (f32 -> f16)
  toh(F[B1_LIN1W], lin1w1h, 512*512);   toh(F[B2_LIN1W], lin1w2h, 512*512);
  toh(F[B2_LIN2W], lin2wh, 128*512);
  toh(F[B1M2_INPROJ], inproj1h, 1024*512); toh(F[B1M2_XPROJ], xproj1h, 65*512);
  toh(F[B1M2_OUTPROJ], outproj1h, 512*512);
  toh(F[B2M2_INPROJ], inproj2h, 256*128);  toh(F[B2M2_XPROJ], xproj2h, 40*128);
  toh(F[B2M2_OUTPROJ], outproj2h, 128*128);
  toh(F[FD1_W1], fd1w1h, 2048*512); toh(F[FD1_W2], fd1w2h, 512*2048);
  toh(F[FD2_W1], fd2w1h, 2048*128); toh(F[FD2_W2], fd2w2h, 128*2048);
  toh(F[LIN3_W], lin3h, 512*128);   toh(F[LIN4_W], lin4h, 96*1024);

  // ---- RevIN + transpose to (M, L)
  revin_kernel<<<M, 256, 0, stream>>>((const float*)d_in[IN_X], F[IN_REVW], F[IN_REVB],
                                      xnT, xnTh, meanv, stdv);

  // ---- block 1: h = lin1(xn)
  gemm(xnTh, lin1w1h, F[B1_LIN1B], h1, 512, nullptr, 0, M, 512, 512, 0);
  // ---- block 2: h = lin2(lin1(xn))
  gemm(xnTh, lin1w2h, F[B2_LIN1B], g2, 512, g2h, 512, M, 512, 512, 0);
  gemm(g2h, lin2wh, F[B2_LIN2B], h2, 128, nullptr, 0, M, 128, 512, 0);

  // ---- decomp + circular correlation -> st (f16, feeds mamba2 in_proj)
  decomp_corr_kernel<<<M, 256, 0, stream>>>(h1, st1h, 512);
  decomp_corr_kernel<<<M, 256, 0, stream>>>(h2, st2h, 128);

  // ---- block1 mambas
  run_mamba2(st1h, 512, 512, 33, 65, inproj1h, F[B1M2_CONVW], F[B1M2_CONVB],
             xproj1h, F[B1M2_DTW], F[B1M2_DTB], F[B1M2_D], outproj1h,
             xz1, xs1, xs1h, xdbl1, dt1, y1h, m2o1);
  run_mamba1(h1, m1o1, 512, B1M1_INPROJ);
  add3_kernel<<<ewg(M * 512), 256, 0, stream>>>(m2o1, m1o1, h1, xo1, M * 512);

  // x1 = LN(xo1)+xo1 ; ffn ; x1 = LN(ffn+x1) -> f16 into cat[:,512:]
  ln(xo1, nullptr, F[LN_G], F[LN_B], x1a, x1ah, 512, 512, 1);
  gemm(x1ah, fd1w1h, F[FD1_B1], nullptr, 0, ffh, 2048, M, 2048, 512, 1);
  gemm(ffh, fd1w2h, F[FD1_B2], f1out, 512, nullptr, 0, M, 512, 2048, 0);
  ln(f1out, x1a, F[LN_G], F[LN_B], nullptr, cat16 + 512, 512, 1024, 0);

  // ---- block2 mambas
  run_mamba2(st2h, 128, 128, 8, 40, inproj2h, F[B2M2_CONVW], F[B2M2_CONVB],
             xproj2h, F[B2M2_DTW], F[B2M2_DTB], F[B2M2_D], outproj2h,
             xz2, xs2, xs2h, xdbl2, dt2, y2h, m2o2);
  run_mamba1(h2, m1o2, 128, B2M1_INPROJ);
  add3_kernel<<<ewg(M * 128), 256, 0, stream>>>(m2o2, m1o2, h2, xo2, M * 128);

  ln(xo2, nullptr, F[LN2_G], F[LN2_B], x2a, x2ah, 128, 128, 1);
  gemm(x2ah, fd2w1h, F[FD2_B1], nullptr, 0, ffh, 2048, M, 2048, 128, 1);
  gemm(ffh, fd2w2h, F[FD2_B2], f2out, 128, nullptr, 0, M, 128, 2048, 0);
  ln(f2out, x2a, F[LN2_G], F[LN2_B], nullptr, x2bh, 128, 128, 0);

  // ---- lin3(x2) -> cat[:, :512] ; lin4(cat) -> outp ; de-normalize
  gemm(x2bh, lin3h, F[LIN3_B], nullptr, 0, cat16, 1024, M, 512, 128, 0);
  gemm(cat16, lin4h, F[LIN4_B], outp, PRED, nullptr, 0, M, PRED, 1024, 0);

  final_kernel<<<ewg(Bn * PRED * Cn), 256, 0, stream>>>(outp, F[IN_REVW], F[IN_REVB],
                                                        meanv, stdv, (float*)d_out);
}
